// CrossAttentionInteraction_40450001994147
// MI455X (gfx1250) — matrix operand order repaired and checked by reading, not compiled, not run on hardware
//
#include <hip/hip_runtime.h>
#include <hip/hip_bf16.h>

typedef unsigned short u16;
typedef __attribute__((ext_vector_type(16))) __bf16 v16bf;
typedef __attribute__((ext_vector_type(8)))  float  v8f;

#define Bt    1024
#define Nn    128
#define Pp    85
#define PPd   96
#define DLIG  256
#define DPROT 1280
#define DATTN 256
#define Hh    4
#define DHd   64
#define FFd   1024

union BF16Frag { v16bf v; u16 s[16]; uint4 q[2]; };
union Pack8    { u16 s[8]; uint4 q; };

__device__ __forceinline__ u16 f2bf(float f) {
  unsigned u = __float_as_uint(f);
  u += 0x7FFFu + ((u >> 16) & 1u);
  return (u16)(u >> 16);
}

// A-matrix 16x32 bf16 fragment; element (m,k) at src[m*ld + k] (k contiguous).
// k order used here (same as B): lanes 0-15 K=0..15, lanes 16-31 K=16..31, contiguous per lane.
__device__ __forceinline__ v16bf load_frag_a(const u16* src, int ld) {
  int lane = threadIdx.x & 31;
  int m = lane & 15, half = lane >> 4;
  const u16* p = src + (size_t)m * ld + half * 16;
  BF16Frag f;
  f.q[0] = *(const uint4*)p;
  f.q[1] = *(const uint4*)(p + 8);
  return f.v;
}

// B-matrix 32x16 bf16 fragment from n-major storage; element (k,n) at src[n*ld + k].
// k order used here (same as A): lanes 0-15 K=0..15, lanes 16-31 K=16..31, contiguous per lane.
__device__ __forceinline__ v16bf load_frag_bT(const u16* src, int ld) {
  int lane = threadIdx.x & 31;
  int n = lane & 15, kh = lane >> 4;
  const u16* p = src + (size_t)n * ld + kh * 16;
  BF16Frag f;
  f.q[0] = ((const uint4*)p)[0];
  f.q[1] = ((const uint4*)p)[1];
  return f.v;
}

__device__ __forceinline__ v8f wmma_bf16(v16bf a, v16bf b, v8f c) {
  return __builtin_amdgcn_wmma_f32_16x16x32_bf16(false, a, false, b, (short)0, c, false, false);
}

// ---------------------------------------------------------------- utility ---
__global__ void k_zero(float* __restrict__ p, int n) {
  int i = blockIdx.x * blockDim.x + threadIdx.x;
  if (i < n) p[i] = 0.f;
}

__global__ void k_cvt_wT(const float* __restrict__ src, u16* __restrict__ dst, int K, int N) {
  int idx = blockIdx.x * blockDim.x + threadIdx.x;
  if (idx < K * N) {
    int k = idx / N, n = idx % N;
    dst[(size_t)n * K + k] = f2bf(src[idx]);
  }
}

// gate bias, ligand/protein valid counts
__global__ void k_prep(const float* __restrict__ conf, const int* __restrict__ pmask,
                       const int* __restrict__ lmask, const float* __restrict__ cw,
                       const float* __restrict__ cb, float* __restrict__ gatebias,
                       float* __restrict__ cnt, float* __restrict__ pcnt) {
  __shared__ float red[128];
  int b = blockIdx.x, t = threadIdx.x;
  float w = cw[0], c0 = cb[0];
  if (t < PPd) {
    float gbv = -__builtin_inff();
    if (t < Pp && pmask[b * Pp + t] == 0) {
      float z = conf[b * Pp + t] * w + c0;
      float sg = 1.f / (1.f + __expf(-z));
      gbv = fmaxf(__logf(sg), -10.f);
    }
    gatebias[b * PPd + t] = gbv;
  }
  red[t] = (t < Nn && lmask[b * Nn + t] != 0) ? 1.f : 0.f;
  __syncthreads();
  for (int s = 64; s; s >>= 1) { if (t < s) red[t] += red[t + s]; __syncthreads(); }
  if (t == 0) cnt[b] = red[0];
  __syncthreads();
  red[t] = (t < Pp && pmask[b * Pp + t] == 0) ? 1.f : 0.f;
  __syncthreads();
  for (int s = 64; s; s >>= 1) { if (t < s) red[t] += red[t + s]; __syncthreads(); }
  if (t == 0) pcnt[b] = red[0];
}

// ------------------------------------------------- LN(ligand) + Q GEMM ------
__global__ __launch_bounds__(128) void k_lnlig_q(
    const float* __restrict__ lig, const int* __restrict__ lmask,
    const float* __restrict__ g, const float* __restrict__ bb,
    const u16* __restrict__ wqT, u16* __restrict__ Q) {
  __shared__ __align__(16) u16 lnA[16][DLIG];
  __shared__ float validS[16];
  int wg = blockIdx.x, b = wg >> 3, r0 = (wg & 7) * 16;
  int wave = threadIdx.x >> 5, lane = threadIdx.x & 31;
  int nl = lane & 15, half = lane >> 4;

  for (int rr = 0; rr < 4; ++rr) {
    int r = wave * 4 + rr;
    const float* row = lig + ((size_t)(b * Nn + r0 + r)) * DLIG;
    float vals[8]; float s = 0.f;
#pragma unroll
    for (int j = 0; j < 8; ++j) { vals[j] = row[lane + 32 * j]; s += vals[j]; }
    for (int m = 16; m; m >>= 1) s += __shfl_xor(s, m, 32);
    float mean = s * (1.f / DLIG);
    float v = 0.f;
#pragma unroll
    for (int j = 0; j < 8; ++j) { float d = vals[j] - mean; v += d * d; }
    for (int m = 16; m; m >>= 1) v += __shfl_xor(v, m, 32);
    float rstd = rsqrtf(v * (1.f / DLIG) + 1e-5f);
#pragma unroll
    for (int j = 0; j < 8; ++j) {
      int c = lane + 32 * j;
      lnA[r][c] = f2bf((vals[j] - mean) * rstd * g[c] + bb[c]);
    }
    if (lane == 0) validS[r] = (lmask[b * Nn + r0 + r] != 0) ? 1.f : 0.f;
  }
  __syncthreads();

  int ct0 = wave * 4;
  v8f z = {};
  v8f acc[4] = { z, z, z, z };
#pragma unroll
  for (int ks = 0; ks < DLIG / 32; ++ks) {
    v16bf a = load_frag_a(&lnA[0][0] + ks * 32, DLIG);
#pragma unroll
    for (int j = 0; j < 4; ++j)
      acc[j] = wmma_bf16(a, load_frag_bT(wqT + (size_t)((ct0 + j) * 16) * DLIG + ks * 32, DLIG), acc[j]);
  }
#pragma unroll
  for (int j = 0; j < 4; ++j) {
    int col = (ct0 + j) * 16 + nl;
#pragma unroll
    for (int i = 0; i < 8; ++i) {
      int m = i + 8 * half;
      Q[((size_t)(b * Nn + r0 + m)) * DATTN + col] = f2bf(acc[j][i] * validS[m]);
    }
  }
}

// ---------------------------------------- LN(protein) + K,V GEMM + sum ------
__global__ __launch_bounds__(128) void k_lnprot_kv(
    const float* __restrict__ prot, const int* __restrict__ pmask,
    const float* __restrict__ g, const float* __restrict__ bb,
    const u16* __restrict__ wkT, const u16* __restrict__ wvT,
    u16* __restrict__ Kb, u16* __restrict__ Vt, float* __restrict__ sprotacc) {
  __shared__ __align__(16) u16 lnP[16][DPROT];   // 40 KB
  int wg = blockIdx.x, b = wg / 6, r0 = (wg % 6) * 16;
  int wave = threadIdx.x >> 5, lane = threadIdx.x & 31;
  int nl = lane & 15, half = lane >> 4;

  for (int rr = 0; rr < 4; ++rr) {
    int r = wave * 4 + rr, pr = r0 + r;
    if (pr < Pp) {
      const float* row = prot + ((size_t)b * Pp + pr) * DPROT;
      bool pvalid = (pmask[b * Pp + pr] == 0);
      float s = 0.f;
      for (int j = 0; j < 40; ++j) {
        float x = row[lane + 32 * j];
        s += x;
        if (pvalid) atomicAdd(&sprotacc[(size_t)b * DPROT + lane + 32 * j], x);
      }
      for (int m = 16; m; m >>= 1) s += __shfl_xor(s, m, 32);
      float mean = s * (1.f / DPROT);
      float v = 0.f;
      for (int j = 0; j < 40; ++j) { float d = row[lane + 32 * j] - mean; v += d * d; }
      for (int m = 16; m; m >>= 1) v += __shfl_xor(v, m, 32);
      float rstd = rsqrtf(v * (1.f / DPROT) + 1e-5f);
      for (int j = 0; j < 40; ++j) {
        int c = lane + 32 * j;
        lnP[r][c] = f2bf((row[c] - mean) * rstd * g[c] + bb[c]);
      }
    } else {
      for (int j = 0; j < 40; ++j) lnP[r][lane + 32 * j] = 0;
    }
  }
  __syncthreads();

  int ct0 = wave * 4;
  v8f z = {};
  v8f ak[4] = { z, z, z, z }, av[4] = { z, z, z, z };
  for (int ks = 0; ks < DPROT / 32; ++ks) {
    v16bf a = load_frag_a(&lnP[0][0] + ks * 32, DPROT);
#pragma unroll
    for (int j = 0; j < 4; ++j) {
      ak[j] = wmma_bf16(a, load_frag_bT(wkT + (size_t)((ct0 + j) * 16) * DPROT + ks * 32, DPROT), ak[j]);
      av[j] = wmma_bf16(a, load_frag_bT(wvT + (size_t)((ct0 + j) * 16) * DPROT + ks * 32, DPROT), av[j]);
    }
  }
#pragma unroll
  for (int j = 0; j < 4; ++j) {
    int col = (ct0 + j) * 16 + nl;
    Pack8 pv;
#pragma unroll
    for (int i = 0; i < 8; ++i) {
      int m = i + 8 * half;
      Kb[((size_t)b * PPd + r0 + m) * DATTN + col] = f2bf(ak[j][i]);
      pv.s[i] = f2bf(av[j][i]);
    }
    // 8 consecutive p values per lane -> one b128 store
    *(uint4*)(Vt + ((size_t)b * DATTN + col) * PPd + r0 + 8 * half) = pv.q;
  }
}

__global__ void k_sprot_mean(const float* __restrict__ acc, const float* __restrict__ pcnt,
                             u16* __restrict__ sprot, int n) {
  int i = blockIdx.x * blockDim.x + threadIdx.x;
  if (i < n) sprot[i] = f2bf(acc[i] / fmaxf(pcnt[i / DPROT], 1.f));
}

// ---------------------------------------------------------- attention -------
__global__ __launch_bounds__(128) void k_attn(
    const u16* __restrict__ Q, const u16* __restrict__ Kb, const u16* __restrict__ Vt,
    const float* __restrict__ gatebias, float* __restrict__ attn_out, u16* __restrict__ ctx) {
  __shared__ __align__(16) u16 attnL[Hh][16][PPd];   // 12 KB
  int wg = blockIdx.x, b = wg >> 3, n0 = (wg & 7) * 16;
  int h = threadIdx.x >> 5, lane = threadIdx.x & 31;
  int nl = lane & 15, half = lane >> 4;

  const u16* Qbase = Q + ((size_t)(b * Nn + n0)) * DATTN + h * DHd;
  const u16* Kbase = Kb + ((size_t)b * PPd) * DATTN + h * DHd;

  v8f z = {};
  v8f acc[6] = { z, z, z, z, z, z };
#pragma unroll
  for (int ks = 0; ks < 2; ++ks) {
    v16bf a = load_frag_a(Qbase + ks * 32, DATTN);
#pragma unroll
    for (int t = 0; t < 6; ++t)
      acc[t] = wmma_bf16(a, load_frag_bT(Kbase + (size_t)(t * 16) * DATTN + ks * 32, DATTN), acc[t]);
  }
  // scale + gate/mask bias
#pragma unroll
  for (int t = 0; t < 6; ++t) {
    float gb = gatebias[(size_t)b * PPd + t * 16 + nl];
#pragma unroll
    for (int i = 0; i < 8; ++i) acc[t][i] = acc[t][i] * 0.125f + gb;
  }
  // softmax over 96 cols (rows live in 16-lane halves)
#pragma unroll
  for (int i = 0; i < 8; ++i) {
    float mx = -__builtin_inff();
#pragma unroll
    for (int t = 0; t < 6; ++t) mx = fmaxf(mx, acc[t][i]);
    for (int s = 8; s; s >>= 1) mx = fmaxf(mx, __shfl_xor(mx, s, 16));
    bool dead = !(mx > -1e37f);
    float sm = 0.f;
#pragma unroll
    for (int t = 0; t < 6; ++t) {
      float e = dead ? 0.f : __expf(acc[t][i] - mx);
      acc[t][i] = e; sm += e;
    }
    for (int s = 8; s; s >>= 1) sm += __shfl_xor(sm, s, 16);
    float inv = (sm > 0.f) ? 1.f / sm : 0.f;
#pragma unroll
    for (int t = 0; t < 6; ++t) acc[t][i] *= inv;
  }
  // write f32 attn to output, bf16 attn to LDS
#pragma unroll
  for (int t = 0; t < 6; ++t) {
    int p = t * 16 + nl;
    if (p < Pp) {
#pragma unroll
      for (int i = 0; i < 8; ++i) {
        int m = i + 8 * half;
        attn_out[(((size_t)(b * Hh + h) * Nn + n0 + m)) * Pp + p] = acc[t][i];
      }
    }
#pragma unroll
    for (int i = 0; i < 8; ++i) attnL[h][i + 8 * half][p] = f2bf(acc[t][i]);
  }
  // ctx = attn @ V   (K-dim padded to 96)
  v8f cacc[4] = { z, z, z, z };
#pragma unroll
  for (int ks = 0; ks < 3; ++ks) {
    v16bf a = load_frag_a(&attnL[h][0][0] + ks * 32, PPd);
#pragma unroll
    for (int ct = 0; ct < 4; ++ct)
      cacc[ct] = wmma_bf16(a,
          load_frag_bT(Vt + ((size_t)b * DATTN + h * DHd + ct * 16) * PPd + ks * 32, PPd), cacc[ct]);
  }
#pragma unroll
  for (int ct = 0; ct < 4; ++ct) {
    int col = h * DHd + ct * 16 + nl;
#pragma unroll
    for (int i = 0; i < 8; ++i)
      ctx[((size_t)(b * Nn + n0 + i + 8 * half)) * DATTN + col] = f2bf(cacc[ct][i]);
  }
}

// ------------------------------------- Wo GEMM + mask + residual + LN -------
__global__ __launch_bounds__(128) void k_wo_ln(
    const u16* __restrict__ ctx, const u16* __restrict__ woT,
    const float* __restrict__ lig, const int* __restrict__ lmask,
    const float* __restrict__ g, const float* __restrict__ bb,
    float* __restrict__ attended) {
  __shared__ float tile[16][DATTN];   // 16 KB
  int wg = blockIdx.x, b = wg >> 3, r0 = (wg & 7) * 16;
  int wave = threadIdx.x >> 5, lane = threadIdx.x & 31;
  int nl = lane & 15, half = lane >> 4;
  const u16* Abase = ctx + ((size_t)(b * Nn + r0)) * DATTN;

  int ct0 = wave * 4;
  v8f z = {};
  v8f acc[4] = { z, z, z, z };
#pragma unroll
  for (int ks = 0; ks < DATTN / 32; ++ks) {
    v16bf a = load_frag_a(Abase + ks * 32, DATTN);
#pragma unroll
    for (int j = 0; j < 4; ++j)
      acc[j] = wmma_bf16(a, load_frag_bT(woT + (size_t)((ct0 + j) * 16) * DATTN + ks * 32, DATTN), acc[j]);
  }
#pragma unroll
  for (int j = 0; j < 4; ++j)
#pragma unroll
    for (int i = 0; i < 8; ++i) tile[i + 8 * half][(ct0 + j) * 16 + nl] = acc[j][i];
  __syncthreads();

  for (int rr = 0; rr < 4; ++rr) {
    int r = wave * 4 + rr;
    size_t row = (size_t)(b * Nn + r0 + r);
    float valid = (lmask[row] != 0) ? 1.f : 0.f;
    const float* lrow = lig + row * DLIG;
    float vals[8]; float s = 0.f;
#pragma unroll
    for (int j = 0; j < 8; ++j) {
      int c = lane + 32 * j;
      float x = tile[r][c] * valid + lrow[c];
      vals[j] = x; s += x;
    }
    for (int m = 16; m; m >>= 1) s += __shfl_xor(s, m, 32);
    float mean = s * (1.f / DATTN);
    float v = 0.f;
#pragma unroll
    for (int j = 0; j < 8; ++j) { float d = vals[j] - mean; v += d * d; }
    for (int m = 16; m; m >>= 1) v += __shfl_xor(v, m, 32);
    float rstd = rsqrtf(v * (1.f / DATTN) + 1e-5f);
#pragma unroll
    for (int j = 0; j < 8; ++j) {
      int c = lane + 32 * j;
      attended[row * DATTN + c] = (vals[j] - mean) * rstd * g[c] + bb[c];
    }
  }
}

// --------------------- FF1 + GELU + FF2 + residual + LN + pool-accum --------
__global__ __launch_bounds__(128) void k_ff(
    const float* __restrict__ attended, const u16* __restrict__ w1T,
    const float* __restrict__ b1, const u16* __restrict__ w2T,
    const float* __restrict__ b2, const float* __restrict__ g,
    const float* __restrict__ bb, const int* __restrict__ lmask,
    float* __restrict__ poolacc) {
  __shared__ __align__(16) u16 attA[16][DATTN];   // 8 KB
  __shared__ __align__(16) u16 hid[16][FFd];      // 32 KB
  __shared__ float tile[16][DATTN];               // 16 KB
  int wg = blockIdx.x, b = wg >> 3, r0 = (wg & 7) * 16;
  int wave = threadIdx.x >> 5, lane = threadIdx.x & 31;
  int nl = lane & 15, half = lane >> 4;
  v8f z = {};

  const float* Af = attended + ((size_t)(b * Nn + r0)) * DATTN;
  for (int idx = threadIdx.x; idx < 16 * DATTN; idx += 128)
    attA[idx >> 8][idx & 255] = f2bf(Af[idx]);
  __syncthreads();

  // FF1: each wave covers 16 column tiles, in 4 groups of 4
  for (int g0 = 0; g0 < 4; ++g0) {
    int ct0 = wave * 16 + g0 * 4;
    v8f acc[4] = { z, z, z, z };
#pragma unroll
    for (int ks = 0; ks < DATTN / 32; ++ks) {
      v16bf a = load_frag_a(&attA[0][0] + ks * 32, DATTN);
#pragma unroll
      for (int j = 0; j < 4; ++j)
        acc[j] = wmma_bf16(a, load_frag_bT(w1T + (size_t)((ct0 + j) * 16) * DATTN + ks * 32, DATTN), acc[j]);
    }
#pragma unroll
    for (int j = 0; j < 4; ++j) {
      int col = (ct0 + j) * 16 + nl;
      float bv = b1[col];
#pragma unroll
      for (int i = 0; i < 8; ++i) {
        float x = acc[j][i] + bv;
        hid[i + 8 * half][col] = f2bf(0.5f * x * (1.f + erff(x * 0.70710678f)));
      }
    }
  }
  __syncthreads();

  // FF2
  {
    int ct0 = wave * 4;
    v8f acc[4] = { z, z, z, z };
    for (int ks = 0; ks < FFd / 32; ++ks) {
      v16bf a = load_frag_a(&hid[0][0] + ks * 32, FFd);
#pragma unroll
      for (int j = 0; j < 4; ++j)
        acc[j] = wmma_bf16(a, load_frag_bT(w2T + (size_t)((ct0 + j) * 16) * FFd + ks * 32, FFd), acc[j]);
    }
#pragma unroll
    for (int j = 0; j < 4; ++j) {
      int col = (ct0 + j) * 16 + nl;
      float bv = b2[col];
#pragma unroll
      for (int i = 0; i < 8; ++i) tile[i + 8 * half][col] = acc[j][i] + bv;
    }
  }
  __syncthreads();

  for (int rr = 0; rr < 4; ++rr) {
    int r = wave * 4 + rr;
    size_t row = (size_t)(b * Nn + r0 + r);
    bool valid = (lmask[row] != 0);
    float vals[8]; float s = 0.f;
#pragma unroll
    for (int j = 0; j < 8; ++j) {
      int c = lane + 32 * j;
      float x = tile[r][c] + attended[row * DATTN + c];
      vals[j] = x; s += x;
    }
    for (int m = 16; m; m >>= 1) s += __shfl_xor(s, m, 32);
    float mean = s * (1.f / DATTN);
    float v = 0.f;
#pragma unroll
    for (int j = 0; j < 8; ++j) { float d = vals[j] - mean; v += d * d; }
    for (int m = 16; m; m >>= 1) v += __shfl_xor(v, m, 32);
    float rstd = rsqrtf(v * (1.f / DATTN) + 1e-5f);
    if (valid) {
#pragma unroll
      for (int j = 0; j < 8; ++j) {
        int c = lane + 32 * j;
        atomicAdd(&poolacc[(size_t)b * DATTN + c], (vals[j] - mean) * rstd * g[c] + bb[c]);
      }
    }
  }
}

// ---------------- protein summary GEMM (pre-pooled means @ Wps + bps) -------
__global__ __launch_bounds__(128) void k_psum(
    const u16* __restrict__ sprot, const u16* __restrict__ wpsT,
    const float* __restrict__ bps, u16* __restrict__ comb) {
  int b0 = blockIdx.x * 16;
  int wave = threadIdx.x >> 5, lane = threadIdx.x & 31;
  int nl = lane & 15, half = lane >> 4;
  const u16* Abase = sprot + (size_t)b0 * DPROT;
  int ct0 = wave * 4;
  v8f z = {};
  v8f acc[4] = { z, z, z, z };
  for (int ks = 0; ks < DPROT / 32; ++ks) {
    v16bf a = load_frag_a(Abase + ks * 32, DPROT);
#pragma unroll
    for (int j = 0; j < 4; ++j)
      acc[j] = wmma_bf16(a, load_frag_bT(wpsT + (size_t)((ct0 + j) * 16) * DPROT + ks * 32, DPROT), acc[j]);
  }
#pragma unroll
  for (int j = 0; j < 4; ++j) {
    int col = (ct0 + j) * 16 + nl;
    float bv = bps[col];
#pragma unroll
    for (int i = 0; i < 8; ++i)
      comb[(size_t)(b0 + i + 8 * half) * 512 + DATTN + col] = f2bf(acc[j][i] + bv);
  }
}

__global__ void k_pool_fin(const float* __restrict__ poolacc, const float* __restrict__ cnt,
                           u16* __restrict__ comb, int n) {
  int i = blockIdx.x * blockDim.x + threadIdx.x;
  if (i < n) {
    int b = i >> 8, c = i & 255;
    comb[(size_t)b * 512 + c] = f2bf(poolacc[i] / fmaxf(cnt[b], 1.f));
  }
}

// ------------------------------------- final concat GEMM (B x 512 x 256) ----
__global__ __launch_bounds__(128) void k_final(
    const u16* __restrict__ comb, const u16* __restrict__ woutT, float* __restrict__ out) {
  int b0 = blockIdx.x * 16;
  int wave = threadIdx.x >> 5, lane = threadIdx.x & 31;
  int nl = lane & 15, half = lane >> 4;
  const u16* Abase = comb + (size_t)b0 * 512;
  int ct0 = wave * 4;
  v8f z = {};
  v8f acc[4] = { z, z, z, z };
#pragma unroll
  for (int ks = 0; ks < 512 / 32; ++ks) {
    v16bf a = load_frag_a(Abase + ks * 32, 512);
#pragma unroll
    for (int j = 0; j < 4; ++j)
      acc[j] = wmma_bf16(a, load_frag_bT(woutT + (size_t)((ct0 + j) * 16) * 512 + ks * 32, 512), acc[j]);
  }
#pragma unroll
  for (int j = 0; j < 4; ++j) {
    int col = (ct0 + j) * 16 + nl;
#pragma unroll
    for (int i = 0; i < 8; ++i)
      out[(size_t)(b0 + i + 8 * half) * DATTN + col] = acc[j][i];
  }
}

// ---------------------------------------------------------------------------
extern "C" void kernel_launch(void* const* d_in, const int* in_sizes, int n_in,
                              void* d_out, int out_size, void* d_ws, size_t ws_size,
                              hipStream_t stream) {
  const float* lig   = (const float*)d_in[0];
  const float* prot  = (const float*)d_in[1];
  const float* conf  = (const float*)d_in[2];
  const int*   pmask = (const int*)d_in[3];
  const int*   lmask = (const int*)d_in[4];
  const float* lng = (const float*)d_in[5],  *lnb = (const float*)d_in[6];
  const float* png = (const float*)d_in[7],  *pnb = (const float*)d_in[8];
  const float* pog = (const float*)d_in[9],  *pob = (const float*)d_in[10];
  const float* ffg = (const float*)d_in[11], *ffb = (const float*)d_in[12];
  const float* Wq = (const float*)d_in[13], *Wk = (const float*)d_in[14];
  const float* Wv = (const float*)d_in[15], *Wo = (const float*)d_in[16];
  const float* cw = (const float*)d_in[17], *cb = (const float*)d_in[18];
  const float* W1 = (const float*)d_in[19], *b1 = (const float*)d_in[20];
  const float* W2 = (const float*)d_in[21], *b2 = (const float*)d_in[22];
  const float* Wps = (const float*)d_in[23], *bps = (const float*)d_in[24];
  const float* Wout = (const float*)d_in[25];

  float* out_vec  = (float*)d_out;
  float* out_attn = out_vec + (size_t)Bt * DATTN;

  char* w = (char*)d_ws; size_t off = 0;
  auto alloc = [&](size_t bytes) -> void* {
    off = (off + 255) & ~(size_t)255;
    void* p = w + off; off += bytes; return p;
  };
  u16*   Q        = (u16*)alloc((size_t)Bt * Nn * DATTN * 2);
  u16*   Kb       = (u16*)alloc((size_t)Bt * PPd * DATTN * 2);
  u16*   Vt       = (u16*)alloc((size_t)Bt * DATTN * PPd * 2);
  u16*   ctx      = (u16*)alloc((size_t)Bt * Nn * DATTN * 2);
  float* attended = (float*)alloc((size_t)Bt * Nn * DATTN * 4);
  float* sprotacc = (float*)alloc((size_t)Bt * DPROT * 4);
  u16*   sprot    = (u16*)alloc((size_t)Bt * DPROT * 2);
  float* poolacc  = (float*)alloc((size_t)Bt * DATTN * 4);
  u16*   comb     = (u16*)alloc((size_t)Bt * 512 * 2);
  float* gatebias = (float*)alloc((size_t)Bt * PPd * 4);
  float* cnt      = (float*)alloc(Bt * 4);
  float* pcnt     = (float*)alloc(Bt * 4);
  u16* wqT  = (u16*)alloc((size_t)DLIG * DATTN * 2);
  u16* wkT  = (u16*)alloc((size_t)DPROT * DATTN * 2);
  u16* wvT  = (u16*)alloc((size_t)DPROT * DATTN * 2);
  u16* woT  = (u16*)alloc((size_t)DATTN * DATTN * 2);
  u16* w1T  = (u16*)alloc((size_t)DATTN * FFd * 2);
  u16* w2T  = (u16*)alloc((size_t)FFd * DATTN * 2);
  u16* wpsT = (u16*)alloc((size_t)DPROT * DATTN * 2);
  u16* woutT= (u16*)alloc((size_t)512 * DATTN * 2);

  // accumulators must be zero every call (deterministic)
  k_zero<<<(Bt * DATTN + 255) / 256, 256, 0, stream>>>(poolacc, Bt * DATTN);
  k_zero<<<(Bt * DPROT + 255) / 256, 256, 0, stream>>>(sprotacc, Bt * DPROT);

  k_prep<<<Bt, 128, 0, stream>>>(conf, pmask, lmask, cw, cb, gatebias, cnt, pcnt);

  auto cvt = [&](const float* src, u16* dst, int K, int N) {
    k_cvt_wT<<<(K * N + 255) / 256, 256, 0, stream>>>(src, dst, K, N);
  };
  cvt(Wq, wqT, DLIG, DATTN);    cvt(Wk, wkT, DPROT, DATTN);
  cvt(Wv, wvT, DPROT, DATTN);   cvt(Wo, woT, DATTN, DATTN);
  cvt(W1, w1T, DATTN, FFd);     cvt(W2, w2T, FFd, DATTN);
  cvt(Wps, wpsT, DPROT, DATTN); cvt(Wout, woutT, 512, DATTN);

  k_lnlig_q<<<Bt * 8, 128, 0, stream>>>(lig, lmask, lng, lnb, wqT, Q);
  k_lnprot_kv<<<Bt * 6, 128, 0, stream>>>(prot, pmask, png, pnb, wkT, wvT, Kb, Vt, sprotacc);
  k_sprot_mean<<<(Bt * DPROT + 255) / 256, 256, 0, stream>>>(sprotacc, pcnt, sprot, Bt * DPROT);

  k_attn<<<Bt * 8, 128, 0, stream>>>(Q, Kb, Vt, gatebias, out_attn, ctx);
  k_wo_ln<<<Bt * 8, 128, 0, stream>>>(ctx, woT, lig, lmask, pog, pob, attended);
  k_ff<<<Bt * 8, 128, 0, stream>>>(attended, w1T, b1, w2T, b2, ffg, ffb, lmask, poolacc);

  k_psum<<<Bt / 16, 128, 0, stream>>>(sprot, wpsT, bps, comb);
  k_pool_fin<<<(Bt * DATTN + 255) / 256, 256, 0, stream>>>(poolacc, cnt, comb, Bt * DATTN);
  k_final<<<Bt / 16, 128, 0, stream>>>(comb, woutT, out_vec);
}
